// MemoryAttention_11158325035072
// MI455X (gfx1250) — matrix operand order repaired and checked by reading, not compiled, not run on hardware
//
#include <hip/hip_runtime.h>
#include <hip/hip_bf16.h>
#include <stdint.h>

// ---------------- problem constants ----------------
#define B_    2
#define S_    4096
#define D_    1024
#define H_    16
#define HD_   64
#define M_    512
#define L_    1024
#define HID_  2816
#define R_    (B_ * M_)     // 1024 rows per step
#define NSTEP (S_ / M_)     // 8 sequential steps
#define EPS_  1e-5f

typedef __bf16 bf16;
typedef __attribute__((ext_vector_type(16))) __bf16 v16bf;
typedef __attribute__((ext_vector_type(8)))  __bf16 v8bf;
typedef __attribute__((ext_vector_type(8)))  float  v8f;

static __device__ __forceinline__ v8f wmma_bf16(v16bf a, v16bf b, v8f c) {
  // D = A(16x32 bf16) * B(32x16 bf16) + C(16x16 f32)
  return __builtin_amdgcn_wmma_f32_16x16x32_bf16(
      false, a, false, b, (short)0, c, false, false);
}

static __device__ __forceinline__ v16bf combine(v8bf lo, v8bf hi) {
  return __builtin_shufflevector(lo, hi, 0, 1, 2, 3, 4, 5, 6, 7, 8, 9, 10, 11,
                                 12, 13, 14, 15);
}

// A-fragment (16x32, per-lane m): two contiguous 8-elem runs at half*8, 16+half*8
static __device__ __forceinline__ v16bf load_a_frag(const bf16* rowbase, int half) {
  v8bf lo = *(const v8bf*)(rowbase + (half << 3));
  v8bf hi = *(const v8bf*)(rowbase + 16 + (half << 3));
  return combine(lo, hi);
}
// B-fragment (32x16, per-lane n) from [n][k]-major storage: two contiguous 8-elem runs at half*8, 16+half*8
static __device__ __forceinline__ v16bf load_b_frag(const bf16* colbase, int half) {
  v8bf lo = *(const v8bf*)(colbase + (half << 3));
  v8bf hi = *(const v8bf*)(colbase + 16 + (half << 3));
  return combine(lo, hi);
}

// LDS aperture keeps the byte offset in addr[31:0]
static __device__ __forceinline__ uint32_t lds_off_u32(const void* p) {
  return (uint32_t)(uintptr_t)p;
}
// gfx1250 async 16B copy global -> LDS, tracked by ASYNCcnt
static __device__ __forceinline__ void async_cp_b128(uint32_t lds_off, const void* g) {
  asm volatile("global_load_async_to_lds_b128 %0, %1, off"
               :: "v"(lds_off), "v"((uint64_t)(uintptr_t)g)
               : "memory");
}
static __device__ __forceinline__ void async_wait0() {
  asm volatile("s_wait_asynccnt 0x0" ::: "memory");
}

// ---------------------------------------------------------------------------
// fp32 -> bf16 convert with batched row gather; optional transpose for weights:
//   src_row = (r / rowsPerBatch) * batchStrideRows + rowOffset + (r % rowsPerBatch)
//   transpose ? dst[c*totalRows + r] : dst[r*cols + c]
// ---------------------------------------------------------------------------
__global__ void conv_gather_bf16(const float* __restrict__ src, bf16* __restrict__ dst,
                                 int rowsPerBatch, int batchStrideRows, int rowOffset,
                                 int cols, int totalRows, int transpose) {
  size_t i = (size_t)blockIdx.x * blockDim.x + threadIdx.x;
  size_t n = (size_t)totalRows * cols;
  if (i >= n) return;
  int r = (int)(i / cols);
  int c = (int)(i % cols);
  int srow = (r / rowsPerBatch) * batchStrideRows + rowOffset + (r % rowsPerBatch);
  bf16 v = (bf16)src[(size_t)srow * cols + c];
  if (transpose) dst[(size_t)c * totalRows + r] = v;
  else           dst[i] = v;
}

// fp32 dense -> fp32 with batched row scatter (final output placement)
__global__ void scatter_f32(const float* __restrict__ src, float* __restrict__ dst,
                            int rowsPerBatch, int batchStrideRows, int rowOffset,
                            int cols, int totalRows) {
  size_t i = (size_t)blockIdx.x * blockDim.x + threadIdx.x;
  size_t n = (size_t)totalRows * cols;
  if (i >= n) return;
  int r = (int)(i / cols);
  int c = (int)(i % cols);
  int drow = (r / rowsPerBatch) * batchStrideRows + rowOffset + (r % rowsPerBatch);
  dst[(size_t)drow * cols + c] = src[i];
}

// ---------------------------------------------------------------------------
// bf16 WMMA GEMM: C(RxN, f32) = A(RxK bf16 row-major) * Wt(NxK bf16, n-major!)
// block tile 64x128, 4 waves, wave tile 32x64 (2x4 WMMA), K-step 32.
// Double-buffered LDS; tiles staged with async b128 copies overlapping compute.
// ---------------------------------------------------------------------------
__global__ __launch_bounds__(128) void gemm_bf16_wmma(
    const bf16* __restrict__ A, const bf16* __restrict__ Wt,
    float* __restrict__ C, int R, int K, int N, int addResidual) {
  __shared__ bf16 As[2][64 * 32];   // [m][k]
  __shared__ bf16 Bs[2][128 * 32];  // [n][k]

  const int tid  = threadIdx.x;
  const int lane = tid & 31;
  const int wid  = tid >> 5;
  const int wm   = wid >> 1;   // 0..1 -> rows wm*32
  const int wn   = wid & 1;    // 0..1 -> cols wn*64
  const int half = lane >> 4;
  const int lm   = lane & 15;
  const int bm   = blockIdx.y * 64;
  const int bn   = blockIdx.x * 128;
  (void)R;

  uint32_t asOff[2] = {lds_off_u32(&As[0][0]), lds_off_u32(&As[1][0])};
  uint32_t bsOff[2] = {lds_off_u32(&Bs[0][0]), lds_off_u32(&Bs[1][0])};

  // stage one 32-wide K tile into LDS buffer `bsel`
  auto stage = [&](int bsel, int k0) {
    // A: 64x32 bf16 = 256 x 16B chunks, 2 per thread
#pragma unroll
    for (int u = 0; u < 2; ++u) {
      int i = tid + u * 128;
      int row = i >> 2, ch = i & 3;
      async_cp_b128(asOff[bsel] + (uint32_t)i * 16,
                    A + (size_t)(bm + row) * K + k0 + ch * 8);
    }
    // Wt: 128x32 bf16 = 512 x 16B chunks, 4 per thread
#pragma unroll
    for (int u = 0; u < 4; ++u) {
      int i = tid + u * 128;
      int row = i >> 2, ch = i & 3;
      async_cp_b128(bsOff[bsel] + (uint32_t)i * 16,
                    Wt + (size_t)(bn + row) * K + k0 + ch * 8);
    }
  };

  v8f acc[2][4] = {};

  stage(0, 0);
  async_wait0();
  __syncthreads();

  int buf = 0;
  for (int k0 = 0; k0 < K; k0 += 32) {
    if (k0 + 32 < K) stage(buf ^ 1, k0 + 32);  // overlap next tile with compute

    v16bf af[2], bfv[4];
#pragma unroll
    for (int i = 0; i < 2; ++i)
      af[i] = load_a_frag(&As[buf][(wm * 32 + i * 16 + lm) * 32], half);
#pragma unroll
    for (int i = 0; i < 4; ++i)
      bfv[i] = load_b_frag(&Bs[buf][(wn * 64 + i * 16 + lm) * 32], half);

#pragma unroll
    for (int i = 0; i < 2; ++i)
#pragma unroll
      for (int jn = 0; jn < 4; ++jn)
        acc[i][jn] = wmma_bf16(af[i], bfv[jn], acc[i][jn]);

    async_wait0();
    __syncthreads();
    buf ^= 1;
  }

  // C layout: row = vgpr r + 8*half, col = lm
  if (addResidual) {
#pragma unroll
    for (int i = 0; i < 2; ++i)
#pragma unroll
      for (int jn = 0; jn < 4; ++jn) {
        int col = bn + wn * 64 + jn * 16 + lm;
#pragma unroll
        for (int r = 0; r < 8; ++r) {
          int row = bm + wm * 32 + i * 16 + r + (half << 3);
          size_t idx = (size_t)row * N + col;
          C[idx] = C[idx] + acc[i][jn][r];
        }
      }
  } else {
#pragma unroll
    for (int i = 0; i < 2; ++i)
#pragma unroll
      for (int jn = 0; jn < 4; ++jn) {
        int col = bn + wn * 64 + jn * 16 + lm;
#pragma unroll
        for (int r = 0; r < 8; ++r) {
          int row = bm + wm * 32 + i * 16 + r + (half << 3);
          C[(size_t)row * N + col] = acc[i][jn][r];
        }
      }
  }
}

// ---------------------------------------------------------------------------
// RMSNorm over last dim (fp32 in, bf16 out). One block (256 thr) per row.
// ---------------------------------------------------------------------------
__global__ __launch_bounds__(256) void rmsnorm_bf16(const float* __restrict__ in,
                                                    const float* __restrict__ w,
                                                    bf16* __restrict__ out, int Dn) {
  __shared__ float red[8];
  int row = blockIdx.x;
  const float* x = in + (size_t)row * Dn;
  float ss = 0.f;
  for (int c = threadIdx.x; c < Dn; c += 256) {
    float v = x[c];
    ss += v * v;
  }
  for (int off = 16; off > 0; off >>= 1) ss += __shfl_xor(ss, off, 32);
  if ((threadIdx.x & 31) == 0) red[threadIdx.x >> 5] = ss;
  __syncthreads();
  float tot = 0.f;
#pragma unroll
  for (int i = 0; i < 8; ++i) tot += red[i];
  float inv = rsqrtf(tot / (float)Dn + EPS_);
  for (int c = threadIdx.x; c < Dn; c += 256)
    out[(size_t)row * Dn + c] = (bf16)(x[c] * inv * w[c]);
}

// silu(g1) * g3 -> bf16
__global__ void silu_mul_bf16(const float* __restrict__ g1, const float* __restrict__ g3,
                              bf16* __restrict__ out, size_t n) {
  size_t i = (size_t)blockIdx.x * blockDim.x + threadIdx.x;
  if (i >= n) return;
  float a = g1[i];
  float s = a / (1.f + __expf(-a));
  out[i] = (bf16)(s * g3[i]);
}

// ---------------------------------------------------------------------------
// Concat [mem | x] along seq, RoPE q/k, pack Q/K as (B,H,L,64) and V as
// (B,H,64,L) bf16 (V dim-major so PV B-fragments are contiguous over keys).
// ---------------------------------------------------------------------------
__global__ void rope_pack_bf16(const float* __restrict__ mk, const float* __restrict__ mv,
                               const float* __restrict__ xq, const float* __restrict__ xk,
                               const float* __restrict__ xv, const float* __restrict__ fcos,
                               const float* __restrict__ fsin, bf16* __restrict__ Qp,
                               bf16* __restrict__ Kp, bf16* __restrict__ Vp) {
  size_t i = (size_t)blockIdx.x * blockDim.x + threadIdx.x;
  if (i >= (size_t)B_ * H_ * L_ * (HD_ / 2)) return;
  int d2 = (int)(i & 31);
  size_t t = i >> 5;
  int pos = (int)(t % L_);
  t /= L_;
  int h = (int)(t % H_);
  int b = (int)(t / H_);
  int bh = b * H_ + h;

  const float *qsrc, *ksrc, *vsrc;
  int srow;
  if (pos < M_) { srow = b * M_ + pos;        qsrc = mk; ksrc = mk; vsrc = mv; }
  else          { srow = b * M_ + (pos - M_); qsrc = xq; ksrc = xk; vsrc = xv; }

  size_t sidx = (size_t)srow * D_ + h * HD_ + d2 * 2;
  float qr = qsrc[sidx], qi = qsrc[sidx + 1];
  float kr = ksrc[sidx], ki = ksrc[sidx + 1];
  float vr = vsrc[sidx], vi = vsrc[sidx + 1];
  float c = fcos[pos * (HD_ / 2) + d2];
  float s = fsin[pos * (HD_ / 2) + d2];

  size_t o = (((size_t)bh * L_ + pos) * HD_) + d2 * 2;
  Qp[o]     = (bf16)(qr * c - qi * s);
  Qp[o + 1] = (bf16)(qr * s + qi * c);
  Kp[o]     = (bf16)(kr * c - ki * s);
  Kp[o + 1] = (bf16)(kr * s + ki * c);
  // V transposed: (B,H,HD,L)
  size_t ov = ((size_t)bh * HD_ + d2 * 2) * L_ + pos;
  Vp[ov]      = (bf16)vr;
  Vp[ov + L_] = (bf16)vi;
}

// ---------------------------------------------------------------------------
// Flash-style causal attention, bf16 WMMA for QK^T and PV, fp32 online softmax.
// grid = (L/64, B*H), block = 128 (4 waves); wave owns 16 query rows.
// Q/K: (B,H,L,64) bf16. V: (B,H,64,L) bf16. O: (B, L, H*64) fp32.
// Double-buffered async K/V staging overlapping softmax + WMMA.
// ---------------------------------------------------------------------------
__global__ __launch_bounds__(128) void attn_wmma(
    const bf16* __restrict__ Q, const bf16* __restrict__ K,
    const bf16* __restrict__ V, float* __restrict__ O, int Hn, int Ln) {
  __shared__ bf16 Ks[2][64 * 64];  // [key][dim]
  __shared__ bf16 Vt[2][64 * 64];  // [dim][key]
  __shared__ bf16 Ps[4][16 * 64];

  const int bh = blockIdx.y;
  const int b = bh / Hn, h = bh % Hn;
  const int q0 = blockIdx.x * 64;
  const bf16* Qp = Q + (size_t)bh * Ln * HD_;
  const bf16* Kp = K + (size_t)bh * Ln * HD_;
  const bf16* Vp = V + (size_t)bh * HD_ * Ln;

  const int tid = threadIdx.x;
  const int lane = tid & 31;
  const int wid = tid >> 5;
  const int half = lane >> 4;
  const int lm = lane & 15;
  uint32_t ksOff[2] = {lds_off_u32(&Ks[0][0]), lds_off_u32(&Ks[1][0])};
  uint32_t vtOff[2] = {lds_off_u32(&Vt[0][0]), lds_off_u32(&Vt[1][0])};

  auto stage = [&](int bsel, int j0) {
    // K tile [key][dim] and V tile [dim][key]: 512 x 16B chunks each, 4/thread
#pragma unroll
    for (int u = 0; u < 4; ++u) {
      int i = tid + u * 128;
      int row = i >> 3, ch = i & 7;
      async_cp_b128(ksOff[bsel] + (uint32_t)i * 16,
                    Kp + (size_t)(j0 + row) * HD_ + ch * 8);
      async_cp_b128(vtOff[bsel] + (uint32_t)i * 16,
                    Vp + (size_t)row * Ln + j0 + ch * 8);
    }
  };

  // Q fragments for this wave's 16 rows (two 32-dim chunks)
  v16bf qf[2];
  {
    int qrow = q0 + wid * 16 + lm;
#pragma unroll
    for (int c = 0; c < 2; ++c)
      qf[c] = load_a_frag(Qp + (size_t)qrow * HD_ + c * 32, half);
  }

  v8f o[4] = {};
  float rm[8], rl[8];
#pragma unroll
  for (int r = 0; r < 8; ++r) { rm[r] = -1e30f; rl[r] = 0.f; }

  const float scale = 0.125f;  // 1/sqrt(64)
  const int jmax = q0 + 64;

  stage(0, 0);
  async_wait0();
  __syncthreads();

  int buf = 0;
  for (int j0 = 0; j0 < jmax; j0 += 64) {
    if (j0 + 64 < jmax) stage(buf ^ 1, j0 + 64);  // prefetch next KV block

    // S strip: 16 q-rows x 64 keys
    v8f s[4];
#pragma unroll
    for (int t = 0; t < 4; ++t) {
      int key = t * 16 + lm;
      v8f sa = {};
      sa = wmma_bf16(qf[0], load_b_frag(&Ks[buf][key * 64 + 0], half), sa);
      sa = wmma_bf16(qf[1], load_b_frag(&Ks[buf][key * 64 + 32], half), sa);
      s[t] = sa;
    }

    // scale + causal mask
#pragma unroll
    for (int t = 0; t < 4; ++t)
#pragma unroll
      for (int r = 0; r < 8; ++r) {
        int qr = q0 + wid * 16 + r + (half << 3);
        int kg = j0 + t * 16 + lm;
        float v = s[t][r] * scale;
        s[t][r] = (kg <= qr) ? v : -3.0e38f;
      }

    // online softmax (reduce across the 16 N-lanes of each half-wave)
#pragma unroll
    for (int r = 0; r < 8; ++r) {
      float m = fmaxf(fmaxf(s[0][r], s[1][r]), fmaxf(s[2][r], s[3][r]));
      for (int x = 1; x < 16; x <<= 1) m = fmaxf(m, __shfl_xor(m, x, 32));
      float mn = fmaxf(rm[r], m);
      float corr = __expf(rm[r] - mn);
      rm[r] = mn;
      rl[r] *= corr;
#pragma unroll
      for (int t = 0; t < 4; ++t) o[t][r] = o[t][r] * corr;
      float rs = 0.f;
#pragma unroll
      for (int t = 0; t < 4; ++t) {
        float e = __expf(s[t][r] - mn);
        s[t][r] = e;
        rs += e;
      }
      for (int x = 1; x < 16; x <<= 1) rs += __shfl_xor(rs, x, 32);
      rl[r] += rs;
    }

    // P (C layout) -> per-wave LDS 16x64 row-major
    bf16* Pw = &Ps[wid][0];
#pragma unroll
    for (int t = 0; t < 4; ++t)
#pragma unroll
      for (int r = 0; r < 8; ++r)
        Pw[(r + (half << 3)) * 64 + t * 16 + lm] = (bf16)s[t][r];
    __syncthreads();

    // O += P * V  (A-frag over keys, contiguous; V B-frag contiguous in [dim][key])
    v16bf pf[2];
#pragma unroll
    for (int c = 0; c < 2; ++c)
      pf[c] = load_a_frag(Pw + lm * 64 + c * 32, half);
#pragma unroll
    for (int t = 0; t < 4; ++t) {
      int dim = t * 16 + lm;
      o[t] = wmma_bf16(pf[0], load_b_frag(&Vt[buf][dim * 64 + 0], half), o[t]);
      o[t] = wmma_bf16(pf[1], load_b_frag(&Vt[buf][dim * 64 + 32], half), o[t]);
    }

    async_wait0();
    __syncthreads();
    buf ^= 1;
  }

  // normalize + store into (B, L, H*64)
#pragma unroll
  for (int t = 0; t < 4; ++t)
#pragma unroll
    for (int r = 0; r < 8; ++r) {
      int qr = q0 + wid * 16 + r + (half << 3);
      int col = h * HD_ + t * 16 + lm;
      O[((size_t)b * Ln + qr) * (size_t)(Hn * HD_) + col] = o[t][r] / rl[r];
    }
}

// ---------------------------------------------------------------------------
// host orchestration
// ---------------------------------------------------------------------------
static inline int blks(size_t n, int t) { return (int)((n + t - 1) / t); }

extern "C" void kernel_launch(void* const* d_in, const int* in_sizes, int n_in,
                              void* d_out, int out_size, void* d_ws, size_t ws_size,
                              hipStream_t stream) {
  (void)in_sizes; (void)n_in; (void)out_size; (void)ws_size;

  const float* x    = (const float*)d_in[0];
  const float* fcos = (const float*)d_in[1];
  const float* fsin = (const float*)d_in[2];
  const float* wq   = (const float*)d_in[3];
  const float* wk   = (const float*)d_in[4];
  const float* wv   = (const float*)d_in[5];
  const float* wo   = (const float*)d_in[6];
  const float* wm   = (const float*)d_in[7];
  const float* wkm  = (const float*)d_in[8];
  const float* wvm  = (const float*)d_in[9];
  const float* w1   = (const float*)d_in[10];
  const float* w3   = (const float*)d_in[11];
  const float* w2   = (const float*)d_in[12];
  const float* ffnw = (const float*)d_in[13];
  const float* memw = (const float*)d_in[14];
  const float* omem = (const float*)d_in[15];
  float* out = (float*)d_out;

  size_t off = 0;
  auto alloc = [&](size_t bytes) -> void* {
    void* p = (char*)d_ws + off;
    off += (bytes + 255) & ~(size_t)255;
    return p;
  };

  // bf16 weights, stored transposed [n][k]
  bf16* wqB  = (bf16*)alloc((size_t)D_ * D_ * 2);
  bf16* wkB  = (bf16*)alloc((size_t)D_ * D_ * 2);
  bf16* wvB  = (bf16*)alloc((size_t)D_ * D_ * 2);
  bf16* woB  = (bf16*)alloc((size_t)D_ * D_ * 2);
  bf16* wmB  = (bf16*)alloc((size_t)D_ * D_ * 2);
  bf16* wkmB = (bf16*)alloc((size_t)D_ * D_ * 2);
  bf16* wvmB = (bf16*)alloc((size_t)D_ * D_ * 2);
  bf16* w1B  = (bf16*)alloc((size_t)D_ * HID_ * 2);
  bf16* w3B  = (bf16*)alloc((size_t)D_ * HID_ * 2);
  bf16* w2B  = (bf16*)alloc((size_t)HID_ * D_ * 2);
  // bf16 activations (row-major; always the A operand)
  bf16* xb   = (bf16*)alloc((size_t)R_ * D_ * 2);
  bf16* omB  = (bf16*)alloc((size_t)R_ * D_ * 2);
  bf16* hb   = (bf16*)alloc((size_t)R_ * D_ * 2);
  bf16* onB  = (bf16*)alloc((size_t)R_ * D_ * 2);
  bf16* gb   = (bf16*)alloc((size_t)R_ * HID_ * 2);
  bf16* QpB  = (bf16*)alloc((size_t)B_ * H_ * L_ * HD_ * 2);
  bf16* KpB  = (bf16*)alloc((size_t)B_ * H_ * L_ * HD_ * 2);
  bf16* VpB  = (bf16*)alloc((size_t)B_ * H_ * HD_ * L_ * 2);
  // fp32 activations
  float* xqF    = (float*)alloc((size_t)R_ * D_ * 4);
  float* xkF    = (float*)alloc((size_t)R_ * D_ * 4);
  float* xvF    = (float*)alloc((size_t)R_ * D_ * 4);
  float* om2F   = (float*)alloc((size_t)R_ * D_ * 4);
  float* g1F    = (float*)alloc((size_t)R_ * HID_ * 4);
  float* g3F    = (float*)alloc((size_t)R_ * HID_ * 4);
  float* mkF    = (float*)alloc((size_t)R_ * D_ * 4);
  float* mvF    = (float*)alloc((size_t)R_ * D_ * 4);
  float* attnO  = (float*)alloc((size_t)B_ * L_ * D_ * 4);
  float* oprojF = (float*)alloc((size_t)R_ * D_ * 4);

  const int CT = 256;
  auto conv = [&](const float* s, bf16* d, int rpb, int bstr, int roff, int cols,
                  int rows, int tr) {
    size_t n = (size_t)rows * cols;
    conv_gather_bf16<<<blks(n, CT), CT, 0, stream>>>(s, d, rpb, bstr, roff, cols,
                                                     rows, tr);
  };
  auto gemm = [&](const bf16* A, const bf16* B, float* C, int R, int K, int N, int res) {
    dim3 g(N / 128, R / 64);
    gemm_bf16_wmma<<<g, 128, 0, stream>>>(A, B, C, R, K, N, res);
  };

  // ---- weight conversion (transposed to [n][k]) ----
  conv(wq,  wqB,  D_,   0, 0, D_,   D_,   1);
  conv(wk,  wkB,  D_,   0, 0, D_,   D_,   1);
  conv(wv,  wvB,  D_,   0, 0, D_,   D_,   1);
  conv(wo,  woB,  D_,   0, 0, D_,   D_,   1);
  conv(wm,  wmB,  D_,   0, 0, D_,   D_,   1);
  conv(wkm, wkmB, D_,   0, 0, D_,   D_,   1);
  conv(wvm, wvmB, D_,   0, 0, D_,   D_,   1);
  conv(w1,  w1B,  D_,   0, 0, HID_, D_,   1);
  conv(w3,  w3B,  D_,   0, 0, HID_, D_,   1);
  conv(w2,  w2B,  HID_, 0, 0, D_,   HID_, 1);
  // origin_mem (1,M,D) broadcast -> (B,M,D): batchStride 0 re-reads rows 0..M-1
  conv(omem, omB, M_, 0, 0, D_, R_, 0);

  for (int s = 0; s < NSTEP; ++s) {
    // x slice -> bf16 (rows b*S + s*M + m)
    conv(x, xb, M_, S_, s * M_, D_, R_, 0);

    gemm(xb, wqB, xqF, R_, D_, D_, 0);
    gemm(xb, wkB, xkF, R_, D_, D_, 0);
    gemm(xb, wvB, xvF, R_, D_, D_, 0);

    // om2 = om @ wm ; om2 += ffn(rmsnorm(om2)) ; om2n = rmsnorm(om2)
    gemm(omB, wmB, om2F, R_, D_, D_, 0);
    rmsnorm_bf16<<<R_, 256, 0, stream>>>(om2F, ffnw, hb, D_);
    gemm(hb, w1B, g1F, R_, D_, HID_, 0);
    gemm(hb, w3B, g3F, R_, D_, HID_, 0);
    silu_mul_bf16<<<blks((size_t)R_ * HID_, CT), CT, 0, stream>>>(
        g1F, g3F, gb, (size_t)R_ * HID_);
    gemm(gb, w2B, om2F, R_, HID_, D_, 1);  // residual add into om2
    rmsnorm_bf16<<<R_, 256, 0, stream>>>(om2F, memw, onB, D_);

    gemm(onB, wkmB, mkF, R_, D_, D_, 0);
    gemm(onB, wvmB, mvF, R_, D_, D_, 0);

    // concat + RoPE + pack (Q,K seq-major; V dim-major)
    rope_pack_bf16<<<blks((size_t)B_ * H_ * L_ * (HD_ / 2), CT), CT, 0, stream>>>(
        mkF, mvF, xqF, xkF, xvF, fcos, fsin, QpB, KpB, VpB);

    // causal attention
    attn_wmma<<<dim3(L_ / 64, B_ * H_), 128, 0, stream>>>(QpB, KpB, VpB, attnO, H_, L_);

    // om_new = attn rows [M, L) -> bf16 (becomes next step's om)
    conv(attnO, omB, M_, L_, M_, D_, R_, 0);

    // project through wo and scatter into d_out rows b*S + s*M + m
    gemm(omB, woB, oprojF, R_, D_, D_, 0);
    scatter_f32<<<blks((size_t)R_ * D_, CT), CT, 0, stream>>>(
        oprojF, out, M_, S_, s * M_, D_, R_);
  }
}
